// Cross_view_STG_50354196578563
// MI455X (gfx1250) — matrix operand order repaired and checked by reading, not compiled, not run on hardware
//
#include <hip/hip_runtime.h>

// ---------------------------------------------------------------- constants
#define T_  4
#define L_  2
#define H_  50000
#define E_  100
#define KPAD 128
#define KC   64

typedef __bf16 v16bf __attribute__((ext_vector_type(16)));
typedef float  v8f   __attribute__((ext_vector_type(8)));

__device__ inline __bf16 f2bf(float f) {
  unsigned u = __builtin_bit_cast(unsigned, f);
  unsigned r = u + 0x7FFFu + ((u >> 16) & 1u);       // round-to-nearest-even
  unsigned short h = (unsigned short)(r >> 16);
  return __builtin_bit_cast(__bf16, h);
}

__device__ inline void atomAddF(float* p, float v) {
  unsafeAtomicAdd(p, v);                              // native global_atomic_add_f32
}

// ---------------------------------------------------------------- WMMA GEMM
// C[n x OUT] = A[n x 100] * B + bias ; B from W (bT=0: W is [100 x OUT] row
// major; bT=1: B = W^T, W is [OUT x 100] row major, i.e. B(k,o)=W[o*100+k]).
//
// Block: 256 threads = 8 waves, 32 output rows x OUTPAD output cols.
// OUTPAD chosen so tiles divide exactly over waves: no EXEC-divergent code
// anywhere around the WMMA (ISA requires EXEC all-ones).
//   wave w: row-tile rt = w>>2 (fixed), col-tiles ct = (w&3)*CPW .. +CPW-1
//   => ONE A-fragment load per k-step, reused for all CPW WMMAs.
template <int OUT>
__global__ __launch_bounds__(256)
void cvstg_gemm_wmma(const float* __restrict__ A, int lda,
                     const float* __restrict__ W, int bT,
                     const float* __restrict__ bias,
                     float* __restrict__ C, int ldc, int nRows)
{
  constexpr int OUTPAD = (OUT <= 128) ? 128 : 320;   // multiple of 64
  constexpr int COLT   = OUTPAD / 16;                // 8 or 20
  constexpr int CPW    = COLT / 4;                   // col tiles per wave: 2 or 5
  constexpr int ROWS   = 32;
  constexpr int SBP    = OUTPAD + 4;                 // LDS padding vs bank conflicts
  constexpr int SAP    = KC + 4;

  __shared__ __bf16 sA[ROWS][SAP];
  __shared__ __bf16 sB[KC][SBP];

  const int tid  = threadIdx.x;
  const int wave = tid >> 5;
  const int lane = tid & 31;
  const int row0 = blockIdx.x * ROWS;

  const int rt  = wave >> 2;                         // wave-fixed row tile
  const int ct0 = (wave & 3) * CPW;                  // first col tile

  v8f acc[CPW] = {};

  for (int kc = 0; kc < KPAD; kc += KC) {
    // stage B chunk (bf16, zero-padded)
    for (int idx = tid; idx < KC * OUTPAD; idx += 256) {
      int k = idx / OUTPAD, o = idx % OUTPAD, kg = kc + k;
      float v = 0.f;
      if (kg < E_ && o < OUT) v = bT ? W[(long)o * E_ + kg] : W[(long)kg * OUT + o];
      sB[k][o] = f2bf(v);
    }
    // stage A chunk
    for (int idx = tid; idx < ROWS * KC; idx += 256) {
      int r = idx / KC, k = idx % KC, row = row0 + r, kg = kc + k;
      float v = (row < nRows && kg < E_) ? A[(long)row * lda + kg] : 0.f;
      sA[r][k] = f2bf(v);
    }
    __syncthreads();

    const int khalf = lane >> 4;
    const int mrow  = lane & 15;
#pragma unroll
    for (int ks = 0; ks < 2; ++ks) {
      v16bf af;                                      // A frag: shared by all CPW tiles
#pragma unroll
      for (int e = 0; e < 16; ++e) {                 // A: lane=M, packed-K layout
        int k = (e < 8) ? (khalf * 8 + e) : (16 + khalf * 8 + (e - 8));
        af[e] = sA[rt * 16 + mrow][ks * 32 + k];
      }
#pragma unroll
      for (int c = 0; c < CPW; ++c) {
        v16bf bfr;
#pragma unroll
        for (int e = 0; e < 16; ++e)                 // B: lane=N col, packed-K layout
          bfr[e] = sB[ks * 32 + ((e < 8) ? (khalf * 8 + e) : (16 + khalf * 8 + (e - 8)))][(ct0 + c) * 16 + mrow];
        acc[c] = __builtin_amdgcn_wmma_f32_16x16x32_bf16(
            false, af, false, bfr, (short)0, acc[c], false, false);
      }
    }
    __syncthreads();
  }

  const int nloc = lane & 15;
  const int half = lane >> 4;
#pragma unroll
  for (int c = 0; c < CPW; ++c) {
    const int col = (ct0 + c) * 16 + nloc;
    if (col < OUT) {
      const float bv = bias ? bias[col] : 0.f;
#pragma unroll
      for (int v = 0; v < 8; ++v) {
        const int row = row0 + rt * 16 + v + 8 * half;
        if (row < nRows) C[(long)row * ldc + col] = acc[c][v] + bv;
      }
    }
  }
}

// ---------------------------------------------------------------- small kernels
__global__ void cvstg_fill(float* p, float v, int n) {
  int i = blockIdx.x * blockDim.x + threadIdx.x;
  if (i < n) p[i] = v;
}

__global__ void cvstg_gcn_deg(float* deg, const int* col, const float* w, int ne) {
  int e = blockIdx.x * blockDim.x + threadIdx.x;
  if (e < ne) atomAddF(&deg[col[e]], w[e]);
}

__global__ void cvstg_rsqrt(float* dinv, const float* deg, int n) {
  int i = blockIdx.x * blockDim.x + threadIdx.x;
  if (i < n) { float d = deg[i]; dinv[i] = d > 0.f ? __frsqrt_rn(d) : 0.f; }
}

__global__ void cvstg_recip(float* inv, const float* d, int n) {
  int i = blockIdx.x * blockDim.x + threadIdx.x;
  if (i < n) { float v = d[i]; inv[i] = v > 0.f ? 1.f / v : 0.f; }
}

__global__ void cvstg_self_init(float* out, const float* xw, const float* dinv, int n) {
  long idx = blockIdx.x * (long)blockDim.x + threadIdx.x;
  if (idx >= (long)n * E_) return;
  long i = idx / E_;
  float di = dinv[i];
  out[idx] = di * di * xw[idx];
}

__global__ void cvstg_gcn_scatter(float* __restrict__ out, const float* __restrict__ xw,
                                  const int* __restrict__ row, const int* __restrict__ col,
                                  const float* __restrict__ w, const float* __restrict__ dinv,
                                  int ne) {
  long idx = blockIdx.x * (long)blockDim.x + threadIdx.x;
  if (idx >= (long)ne * (E_ / 4)) return;
  int e = (int)(idx / (E_ / 4)), f = (int)(idx % (E_ / 4)) * 4;
  int r = row[e], c = col[e];
  float nm = dinv[r] * w[e] * dinv[c];
  float4 xv = *(const float4*)(xw + (long)r * E_ + f);
  float* op = out + (long)c * E_ + f;
  atomAddF(op + 0, nm * xv.x); atomAddF(op + 1, nm * xv.y);
  atomAddF(op + 2, nm * xv.z); atomAddF(op + 3, nm * xv.w);
}

__global__ void cvstg_gcn_finalize(float* __restrict__ Gw, const float* __restrict__ outb,
                                   const float* __restrict__ bias, int t, int n) {
  int wid  = (int)((blockIdx.x * (long)blockDim.x + threadIdx.x) >> 5);
  int lane = threadIdx.x & 31;
  if (wid >= n) return;                       // wave-uniform exit
  float vals[4]; float ss = 0.f;
#pragma unroll
  for (int j = 0; j < 4; ++j) {
    int f = lane + 32 * j; float v = 0.f;
    if (f < E_) v = outb[(long)wid * E_ + f] + bias[f];
    vals[j] = v; ss += v * v;
  }
#pragma unroll
  for (int off = 16; off >= 1; off >>= 1) ss += __shfl_xor(ss, off, 32);
  float rn = 1.f / fmaxf(sqrtf(ss), 1e-12f);
#pragma unroll
  for (int j = 0; j < 4; ++j) {
    int f = lane + 32 * j;
    if (f < E_) Gw[((long)wid * T_ + t) * E_ + f] += vals[j] * rn;
  }
}

__global__ void cvstg_hyp_count(float* D, float* B, const int* node, const int* he, int m) {
  int i = blockIdx.x * blockDim.x + threadIdx.x;
  if (i < m) { atomAddF(&D[node[i]], 1.f); atomAddF(&B[he[i]], 1.f); }
}

__global__ void cvstg_hyp_gather(float* __restrict__ mbuf, const float* __restrict__ xw,
                                 const int* __restrict__ node, const int* __restrict__ he, int m) {
  long idx = blockIdx.x * (long)blockDim.x + threadIdx.x;
  if (idx >= (long)m * (E_ / 4)) return;
  int i = (int)(idx / (E_ / 4)), f = (int)(idx % (E_ / 4)) * 4;
  float4 xv = *(const float4*)(xw + (long)node[i] * E_ + f);
  float* op = mbuf + (long)he[i] * E_ + f;
  atomAddF(op + 0, xv.x); atomAddF(op + 1, xv.y);
  atomAddF(op + 2, xv.z); atomAddF(op + 3, xv.w);
}

__global__ void cvstg_scale_rows(float* m, const float* inv, int rows) {
  long idx = blockIdx.x * (long)blockDim.x + threadIdx.x;
  if (idx < (long)rows * E_) m[idx] *= inv[idx / E_];
}

__global__ void cvstg_hyp_scatter(float* __restrict__ out, const float* __restrict__ mbuf,
                                  const int* __restrict__ node, const int* __restrict__ he, int m) {
  long idx = blockIdx.x * (long)blockDim.x + threadIdx.x;
  if (idx >= (long)m * (E_ / 4)) return;
  int i = (int)(idx / (E_ / 4)), f = (int)(idx % (E_ / 4)) * 4;
  float4 mv = *(const float4*)(mbuf + (long)he[i] * E_ + f);
  float* op = out + (long)node[i] * E_ + f;
  atomAddF(op + 0, mv.x); atomAddF(op + 1, mv.y);
  atomAddF(op + 2, mv.z); atomAddF(op + 3, mv.w);
}

__global__ void cvstg_hyp_finalize(float* __restrict__ Hw, const float* __restrict__ outb,
                                   const float* __restrict__ Dinv,
                                   const float* __restrict__ bias, int t, int n) {
  int wid  = (int)((blockIdx.x * (long)blockDim.x + threadIdx.x) >> 5);
  int lane = threadIdx.x & 31;
  if (wid >= n) return;
  float di = Dinv[wid];
  float vals[4]; float ss = 0.f;
#pragma unroll
  for (int j = 0; j < 4; ++j) {
    int f = lane + 32 * j; float v = 0.f;
    if (f < E_) v = outb[(long)wid * E_ + f] * di + bias[f];
    vals[j] = v; ss += v * v;
  }
#pragma unroll
  for (int off = 16; off >= 1; off >>= 1) ss += __shfl_xor(ss, off, 32);
  float rn = 1.f / fmaxf(sqrtf(ss), 1e-12f);
#pragma unroll
  for (int j = 0; j < 4; ++j) {
    int f = lane + 32 * j;
    if (f < E_) Hw[((long)wid * T_ + t) * E_ + f] += vals[j] * rn;
  }
}

__global__ void cvstg_gru_gates(float* __restrict__ h, float* __restrict__ hall,
                                const float* __restrict__ gi, const float* __restrict__ gh,
                                int t, int n) {
  long idx = blockIdx.x * (long)blockDim.x + threadIdx.x;
  if (idx >= (long)n * E_) return;
  long i = idx / E_; int f = (int)(idx % E_);
  const float* gir = gi + i * (3 * E_);
  const float* ghr = gh + i * (3 * E_);
  float r  = 1.f / (1.f + __expf(-(gir[f] + ghr[f])));
  float z  = 1.f / (1.f + __expf(-(gir[E_ + f] + ghr[E_ + f])));
  float ng = tanhf(gir[2 * E_ + f] + r * ghr[2 * E_ + f]);
  float hp = h[idx];
  float hn = (1.f - z) * ng + z * hp;
  h[idx]   = hn;
  hall[(i * T_ + t) * E_ + f] = hn;
}

// ---------------------------------------------------------------- host
extern "C" void kernel_launch(void* const* d_in, const int* in_sizes, int n_in,
                              void* d_out, int out_size, void* d_ws, size_t ws_size,
                              hipStream_t stream) {
  const int N  = in_sizes[0] / E_;
  const int NE = in_sizes[2] / T_;
  const int M  = in_sizes[3] / T_;

  const float* emb  = (const float*)d_in[0];
  const int*   gei  = (const int*)d_in[1];
  const float* gew  = (const float*)d_in[2];
  const int*   hynd = (const int*)d_in[3];
  const int*   hyed = (const int*)d_in[4];
  const float* Wg   = (const float*)d_in[5];
  const float* bg   = (const float*)d_in[6];
  const float* Wh   = (const float*)d_in[7];
  const float* bh   = (const float*)d_in[8];

  const size_t nE = (size_t)N * E_ * sizeof(float);
  char* ws = (char*)d_ws;
  float* Gw    = (float*)ws;                          // [N,T,E] (reused as Hw)
  char*  sp    = ws + (size_t)T_ * nE;                // scratch pool
  float* xw    = (float*)sp;                          // conv: N*E
  float* outb  = (float*)(sp + nE);                   // conv: N*E
  float* mbuf  = (float*)(sp + 2 * nE);               // conv: H*E
  float* gi    = (float*)sp;                          // GRU: N*3E (overlaps conv scratch)
  float* gh    = (float*)(sp + 3 * nE);               // GRU: N*3E
  char*  tail  = sp + 6 * nE;
  float* hbuf  = (float*)tail;                        // N*E
  float* deg   = (float*)(tail + nE);                 // N
  float* dinv  = deg + N;                             // N
  float* Dninv = dinv + N;                            // N
  float* Bdeg  = Dninv + N;                           // H
  float* Binv  = Bdeg + H_;                           // H

  float* x1 = (float*)d_out;
  float* x2 = x1 + (size_t)N * E_;
  float* h1 = x2 + (size_t)N * E_;
  float* h2 = h1 + (size_t)N * T_ * E_;

  const int TB = 256;
  const int gN    = (N + TB - 1) / TB;
  const int gNE_  = (N * E_ + TB - 1) / TB;
  const int gEdge = (NE * (E_ / 4) + TB - 1) / TB;
  const int gNnz  = (M * (E_ / 4) + TB - 1) / TB;
  const int gWaveN = (N + 7) / 8;                     // 8 waves/block, 1 node/wave
  const int gGemm  = (N + 31) / 32;

  // ---------------- Phase 1: GCN stack -> Gw
  hipMemsetAsync(Gw, 0, (size_t)T_ * nE, stream);
  for (int t = 0; t < T_; ++t) {
    const int* row = gei + (size_t)t * 2 * NE;
    const int* col = row + NE;
    const float* w = gew + (size_t)t * NE;
    cvstg_fill<<<gN, TB, 0, stream>>>(deg, 1.0f, N);  // self-loop weight
    cvstg_gcn_deg<<<(NE + TB - 1) / TB, TB, 0, stream>>>(deg, col, w, NE);
    cvstg_rsqrt<<<gN, TB, 0, stream>>>(dinv, deg, N);
    for (int l = 0; l < L_; ++l) {
      const float* Wtl = Wg + ((size_t)t * L_ + l) * E_ * E_;
      const float* btl = bg + ((size_t)t * L_ + l) * E_;
      cvstg_gemm_wmma<E_><<<gGemm, TB, 0, stream>>>(emb, E_, Wtl, 0, nullptr, xw, E_, N);
      cvstg_self_init<<<gNE_, TB, 0, stream>>>(outb, xw, dinv, N);
      cvstg_gcn_scatter<<<gEdge, TB, 0, stream>>>(outb, xw, row, col, w, dinv, NE);
      cvstg_gcn_finalize<<<gWaveN, TB, 0, stream>>>(Gw, outb, btl, t, N);
    }
  }

  // ---------------- GRU 1
  hipMemsetAsync(hbuf, 0, nE, stream);
  for (int t = 0; t < T_; ++t) {
    cvstg_gemm_wmma<3 * E_><<<gGemm, TB, 0, stream>>>(
        Gw + (size_t)t * E_, T_ * E_, (const float*)d_in[9], 1,
        (const float*)d_in[11], gi, 3 * E_, N);
    cvstg_gemm_wmma<3 * E_><<<gGemm, TB, 0, stream>>>(
        hbuf, E_, (const float*)d_in[10], 1, (const float*)d_in[12], gh, 3 * E_, N);
    cvstg_gru_gates<<<gNE_, TB, 0, stream>>>(hbuf, h1, gi, gh, t, N);
  }
  hipMemcpyAsync(x1, hbuf, nE, hipMemcpyDeviceToDevice, stream);

  // ---------------- Phase 2: Hypergraph stack -> Gw (reused as Hw)
  hipMemsetAsync(Gw, 0, (size_t)T_ * nE, stream);
  for (int t = 0; t < T_; ++t) {
    const int* node = hynd + (size_t)t * M;
    const int* he   = hyed + (size_t)t * M;
    hipMemsetAsync(deg, 0, (size_t)N * sizeof(float), stream);
    hipMemsetAsync(Bdeg, 0, (size_t)H_ * sizeof(float), stream);
    cvstg_hyp_count<<<(M + TB - 1) / TB, TB, 0, stream>>>(deg, Bdeg, node, he, M);
    cvstg_recip<<<gN, TB, 0, stream>>>(Dninv, deg, N);
    cvstg_recip<<<(H_ + TB - 1) / TB, TB, 0, stream>>>(Binv, Bdeg, H_);
    for (int l = 0; l < L_; ++l) {
      const float* Wtl = Wh + ((size_t)t * L_ + l) * E_ * E_;
      const float* btl = bh + ((size_t)t * L_ + l) * E_;
      cvstg_gemm_wmma<E_><<<gGemm, TB, 0, stream>>>(emb, E_, Wtl, 0, nullptr, xw, E_, N);
      hipMemsetAsync(mbuf, 0, (size_t)H_ * E_ * sizeof(float), stream);
      cvstg_hyp_gather<<<gNnz, TB, 0, stream>>>(mbuf, xw, node, he, M);
      cvstg_scale_rows<<<(H_ * E_ + TB - 1) / TB, TB, 0, stream>>>(mbuf, Binv, H_);
      hipMemsetAsync(outb, 0, nE, stream);
      cvstg_hyp_scatter<<<gNnz, TB, 0, stream>>>(outb, mbuf, node, he, M);
      cvstg_hyp_finalize<<<gWaveN, TB, 0, stream>>>(Gw, outb, Dninv, btl, t, N);
    }
  }

  // ---------------- GRU 2
  hipMemsetAsync(hbuf, 0, nE, stream);
  for (int t = 0; t < T_; ++t) {
    cvstg_gemm_wmma<3 * E_><<<gGemm, TB, 0, stream>>>(
        Gw + (size_t)t * E_, T_ * E_, (const float*)d_in[13], 1,
        (const float*)d_in[15], gi, 3 * E_, N);
    cvstg_gemm_wmma<3 * E_><<<gGemm, TB, 0, stream>>>(
        hbuf, E_, (const float*)d_in[14], 1, (const float*)d_in[16], gh, 3 * E_, N);
    cvstg_gru_gates<<<gNE_, TB, 0, stream>>>(hbuf, h2, gi, gh, t, N);
  }
  hipMemcpyAsync(x2, hbuf, nE, hipMemcpyDeviceToDevice, stream);

  (void)n_in; (void)out_size; (void)ws_size;
}
